// branch_conv_24876450578678
// MI455X (gfx1250) — matrix operand order repaired and checked by reading, not compiled, not run on hardware
//
#include <hip/hip_runtime.h>
#include <hip/hip_bf16.h>
#include <cstdint>
#include <cstddef>

// ---------------- CDNA5 WMMA types ----------------
typedef __attribute__((ext_vector_type(16))) _Float16 v16h;
typedef __attribute__((ext_vector_type(8)))  _Float16 v8h;
typedef __attribute__((ext_vector_type(8)))  float    v8f;
typedef __attribute__((ext_vector_type(4)))  unsigned int u4;

union F16x16 { v16h v; u4 u[2]; };
union F16x8  { v8h  v; u4 u;   };

// ---------------- problem constants ----------------
#define NBATCH 8
#define CIN    256
#define OCH    256
#define HH     64
#define WW     64
#define HWSZ   4096          // HH*WW
#define KK     9
#define KDIM   2304          // CIN*KK
#define NQ     72            // KDIM/32 (K-chunks per WMMA)
#define NPOS   32768         // NBATCH*HWSZ

// A-fragment: 16x32 f16, lane L: M=L%16, halves at K = h*8.. and 16+h*8..
__device__ inline v16h load_a_frag(const _Float16* p0) {
  F16x16 r;
  r.u[0] = *(const u4*)(p0);
  r.u[1] = *(const u4*)(p0 + 16);
  return r.v;
}
// B-fragment pre-packed: 16 contiguous halves per lane
__device__ inline v16h load_b_frag(const _Float16* p) {
  F16x16 r;
  r.u[0] = *(const u4*)(p);
  r.u[1] = *(const u4*)(p + 8);
  return r.v;
}

// ---------------- K1: x NCHW f32 -> NHWC f16 ----------------
__global__ __launch_bounds__(256) void k_x_to_nhwc(const float* __restrict__ x,
                                                   _Float16* __restrict__ xh) {
  int idx = blockIdx.x * 256 + threadIdx.x;     // idx = pos*256 + c
  int c   = idx & 255;
  int pos = idx >> 8;                           // n*4096 + sp
  int n   = pos >> 12;
  int sp  = pos & 4095;
  xh[idx] = (_Float16)x[((size_t)(n * CIN + c)) * HWSZ + sp];
}

// ---------------- K2: pack weights into per-lane B-fragment order ----------
// layout: frag[((nt*NQ + q)*32 + lane)*16 + p] = W[K = q*32 + (lane/16)*8 + p + (p>=8 ? 8 : 0)][oc = nt*16 + lane%16]
// with K index = k*256 + c and W[k*256+c][oc] = weight_OIHW[oc][c][k]
__global__ __launch_bounds__(256) void k_prep_w(const float* __restrict__ w,
                                                const float* __restrict__ ow,
                                                _Float16* __restrict__ wf,
                                                _Float16* __restrict__ owf) {
  int tid = blockIdx.x * 256 + threadIdx.x;
  const int MAIN = 16 * NQ * 32;                 // 36864 frags (main weight)
  const int OFFW = 2 * NQ * 32;                  // 4608 frags (offset weight)
  if (tid >= MAIN + OFFW) return;
  bool is_off = tid >= MAIN;
  int t = is_off ? (tid - MAIN) : tid;
  int nt   = t / (NQ * 32);
  int r    = t % (NQ * 32);
  int q    = r >> 5;
  int lane = r & 31;
  int oc   = nt * 16 + (lane & 15);
  int kb   = q * 32 + (lane >> 4) * 8;
  _Float16* dst = (is_off ? owf : wf) + (size_t)t * 16;
#pragma unroll
  for (int p = 0; p < 16; ++p) {
    int kidx = kb + p + (p >= 8 ? 8 : 0);
    int k = kidx >> 8;          // tap 0..8
    int c = kidx & 255;         // channel
    float v;
    if (is_off) v = (oc < 27) ? ow[((size_t)(oc * CIN + c)) * KK + k] : 0.0f;
    else        v = w[((size_t)(oc * CIN + c)) * KK + k];
    dst[p] = (_Float16)v;
  }
}

// ---------------- K3: offset conv (implicit GEMM, WMMA) --------------------
// M-tile 32 positions, N = 32 (27 valid), 128 threads = 4 waves = 2 Msub x 2 Ntile
// output: offs[p*32 + ch], ch 0..17 raw offsets (+bias), 18..26 sigmoid mask
__global__ __launch_bounds__(128) void k_offset_conv(const _Float16* __restrict__ xh,
                                                     const _Float16* __restrict__ owf,
                                                     const float* __restrict__ obias,
                                                     float* __restrict__ offs) {
  __shared__ _Float16 ldsA[32 * 256];
  const int t     = threadIdx.x;
  const int pbase = blockIdx.x * 32;
  const int lane  = t & 31;
  const int wv    = t >> 5;
  const int m_sub = wv >> 1;
  const int nt    = wv & 1;
  v8f acc = {};

  // builder mapping: thread -> (position, 64-channel slab)
  const int bm = t >> 2;
  const int bc = (t & 3) << 6;
  const int p  = pbase + bm;
  const int n_img = p >> 12;
  const int sp    = p & 4095;
  const int ho = sp >> 6, wo = sp & 63;

  const u4 zero = {0u, 0u, 0u, 0u};

  for (int k = 0; k < KK; ++k) {
    int kh = k / 3, kw = k % 3;
    int hi = ho - 1 + kh, wi = wo - 1 + kw;
    bool valid = (hi >= 0) & (hi < HH) & (wi >= 0) & (wi < WW);
    int hc = min(HH - 1, max(0, hi));
    int wc = min(WW - 1, max(0, wi));
    const u4* src = (const u4*)(xh + ((((size_t)n_img * HH + hc) * WW + wc) << 8) + bc);
    u4* dst = (u4*)(ldsA + bm * 256 + bc);
#pragma unroll
    for (int j = 0; j < 8; ++j) {
      u4 v = src[j];
      if (!valid) v = zero;
      dst[j] = v;
    }
    __syncthreads();
#pragma unroll
    for (int ql = 0; ql < 8; ++ql) {
      int q = k * 8 + ql;
      const _Float16* a0 = ldsA + (m_sub * 16 + (lane & 15)) * 256 + ql * 32 + (lane >> 4) * 8;
      v16h a = load_a_frag(a0);
      v16h b = load_b_frag(owf + ((size_t)(nt * NQ + q) * 32 + lane) * 16);
      acc = __builtin_amdgcn_wmma_f32_16x16x32_f16(false, a, false, b, (short)0, acc, false, false);
    }
    __syncthreads();
  }

  int oc = nt * 16 + (lane & 15);
  if (oc < 27) {
    float bv = obias[oc];
    int half = lane >> 4;
#pragma unroll
    for (int i = 0; i < 8; ++i) {
      int pp = pbase + m_sub * 16 + i + half * 8;
      float v = acc[i] + bv;
      if (oc >= 18) v = 1.0f / (1.0f + __expf(-v));   // sigmoid mask
      offs[(size_t)pp * 32 + oc] = v;
    }
  }
}

// ---------------- K4: main deformable conv (implicit GEMM, WMMA) -----------
// M-tile 32 positions, N = 256, 256 threads = 8 waves = 2 Msub x 4 Ngroup(4 tiles)
__global__ __launch_bounds__(256) void k_main_conv(const _Float16* __restrict__ xh,
                                                   const _Float16* __restrict__ wf,
                                                   const float* __restrict__ offs,
                                                   const float* __restrict__ bias,
                                                   float* __restrict__ out) {
  __shared__ _Float16 ldsA[32 * 256];
  const int t     = threadIdx.x;
  const int pbase = blockIdx.x * 32;
  const int lane  = t & 31;
  const int wv    = t >> 5;
  const int m_sub = wv >> 2;     // 0..1
  const int ng    = wv & 3;      // 0..3 -> 4 N-tiles each
  v8f acc[4] = {};

  // builder mapping: thread -> (position, 32-channel slab)
  const int bm = t >> 3;          // 0..31
  const int bc = (t & 7) << 5;    // 0..224
  const int p  = pbase + bm;
  const int n_img = p >> 12;
  const int sp    = p & 4095;
  const int ho = sp >> 6, wo = sp & 63;
  const float* op = offs + (size_t)p * 32;

  for (int k = 0; k < KK; ++k) {
    int kh = k / 3, kw = k % 3;
    // --- bilinear sample 32 channels for this tap, fold sigmoid mask in ---
    float dy = op[2 * k], dx = op[2 * k + 1], mk = op[18 + k];
    float h = (float)(ho - 1 + kh) + dy;
    float w = (float)(wo - 1 + kw) + dx;
    float h0f = floorf(h), w0f = floorf(w);
    float lh = h - h0f, lw = w - w0f;
    int h0 = (int)h0f, w0 = (int)w0f;
    float wgt[4];
    wgt[0] = (1.0f - lh) * (1.0f - lw) * mk;
    wgt[1] = (1.0f - lh) * lw * mk;
    wgt[2] = lh * (1.0f - lw) * mk;
    wgt[3] = lh * lw * mk;
    const _Float16* nb[4];
#pragma unroll
    for (int d = 0; d < 4; ++d) {
      int hi = h0 + (d >> 1);
      int wi = w0 + (d & 1);
      bool valid = (hi >= 0) & (hi < HH) & (wi >= 0) & (wi < WW);
      if (!valid) wgt[d] = 0.0f;
      int hc = min(HH - 1, max(0, hi));
      int wc = min(WW - 1, max(0, wi));
      nb[d] = xh + ((((size_t)n_img * HH + hc) * WW + wc) << 8) + bc;
    }
#pragma unroll
    for (int cc = 0; cc < 4; ++cc) {            // 4 sub-chunks of 8 channels
      float a[8] = {0, 0, 0, 0, 0, 0, 0, 0};
#pragma unroll
      for (int d = 0; d < 4; ++d) {
        F16x8 v;
        v.u = *(const u4*)(nb[d] + cc * 8);
#pragma unroll
        for (int j = 0; j < 8; ++j) a[j] += wgt[d] * (float)v.v[j];
      }
      F16x8 o;
#pragma unroll
      for (int j = 0; j < 8; ++j) o.v[j] = (_Float16)a[j];
      *(u4*)(ldsA + bm * 256 + bc + cc * 8) = o.u;
    }
    __syncthreads();
    // --- WMMA over the 8 K-chunks of this tap ---
#pragma unroll
    for (int ql = 0; ql < 8; ++ql) {
      int q = k * 8 + ql;
      const _Float16* a0 = ldsA + (m_sub * 16 + (lane & 15)) * 256 + ql * 32 + (lane >> 4) * 8;
      v16h afrag = load_a_frag(a0);
#pragma unroll
      for (int j = 0; j < 4; ++j) {
        v16h bfrag = load_b_frag(wf + ((size_t)((ng * 4 + j) * NQ + q) * 32 + lane) * 16);
        acc[j] = __builtin_amdgcn_wmma_f32_16x16x32_f16(false, afrag, false, bfrag,
                                                        (short)0, acc[j], false, false);
      }
    }
    __syncthreads();
  }

  // --- epilogue: +bias, scatter to NCHW f32 out ---
  const int nl   = lane & 15;
  const int half = lane >> 4;
#pragma unroll
  for (int j = 0; j < 4; ++j) {
    int oc = (ng * 4 + j) * 16 + nl;
    float bv = bias[oc];
#pragma unroll
    for (int i = 0; i < 8; ++i) {
      int pp  = pbase + m_sub * 16 + i + half * 8;
      int ni  = pp >> 12;
      int spp = pp & 4095;
      out[((size_t)(ni * OCH + oc)) * HWSZ + spp] = acc[j][i] + bv;
    }
  }
}

// ---------------- launch ----------------
extern "C" void kernel_launch(void* const* d_in, const int* in_sizes, int n_in,
                              void* d_out, int out_size, void* d_ws, size_t ws_size,
                              hipStream_t stream) {
  const float* x      = (const float*)d_in[1];
  const float* weight = (const float*)d_in[2];
  const float* bias   = (const float*)d_in[3];
  const float* ow     = (const float*)d_in[4];
  const float* obias  = (const float*)d_in[5];
  float* out = (float*)d_out;

  char* ws = (char*)d_ws;
  const size_t XH_BYTES  = (size_t)NPOS * CIN * 2;           // 16 MiB  NHWC f16
  const size_t WF_BYTES  = (size_t)16 * NQ * 32 * 16 * 2;    // 1.125 MiB packed B
  const size_t OWF_BYTES = (size_t)2 * NQ * 32 * 16 * 2;     // 144 KiB packed B
  _Float16* xh   = (_Float16*)(ws);
  _Float16* wf   = (_Float16*)(ws + XH_BYTES);
  _Float16* owf  = (_Float16*)(ws + XH_BYTES + WF_BYTES);
  float*    offs = (float*)   (ws + XH_BYTES + WF_BYTES + OWF_BYTES); // 4 MiB

  // K1: transpose/downcast x
  k_x_to_nhwc<<<(NPOS * CIN) / 256, 256, 0, stream>>>(x, xh);
  // K2: pack both weight tensors into WMMA B-fragment order
  k_prep_w<<<(16 * NQ * 32 + 2 * NQ * 32 + 255) / 256, 256, 0, stream>>>(weight, ow, wf, owf);
  // K3: offset conv -> dy/dx/sigmoid(mask)
  k_offset_conv<<<NPOS / 32, 128, 0, stream>>>(xh, owf, obias, offs);
  // K4: deformable main conv
  k_main_conv<<<NPOS / 32, 256, 0, stream>>>(xh, wf, offs, bias, out);
}
